// MemoryModule_37529424233068
// MI455X (gfx1250) — matrix operand order repaired and checked by reading, not compiled, not run on hardware
//
#include <hip/hip_runtime.h>
#include <hip/hip_bf16.h>

typedef __attribute__((ext_vector_type(16))) _Float16 v16h;
typedef __attribute__((ext_vector_type(8)))  float    v8f;

#define WMMA_F16(a,b,c) __builtin_amdgcn_wmma_f32_16x16x32_f16(false,(a),false,(b),(short)0,(c),false,false)

#define NE        100000
#define NNODES    20000
#define MEM       256
#define EDGE      128
#define TIMED     64
#define MSG       256
#define IN_DIM    704      // 256+256+128+64
#define G3        768      // 3*MEM
#define MTILES_E  6250     // NE/16
#define MTILES_N  1250     // NNODES/16

// ---------- helpers ----------

// A-matrix 16x32 f16 fragment K offset for vgpr-pair v, lane-half h (ISA 7.12.2)
__device__ __forceinline__ int a_koff(int v, int half) {
  return ((v < 4) ? (2 * v) : (16 + 2 * (v - 4))) + 8 * half;
}

// Build A fragment from an f32 row (row-major, contiguous K), local K start kloc
__device__ __forceinline__ v16h a_frag_f32(const float* __restrict__ row, int kloc, int half) {
  v16h a;
#pragma unroll
  for (int v = 0; v < 8; ++v) {
    int k = kloc + a_koff(v, half);
    float2 f = *(const float2*)(row + k);
    a[2 * v]     = (_Float16)f.x;
    a[2 * v + 1] = (_Float16)f.y;
  }
  return a;
}

// Build A fragment for the time-encoding segment: cos(rt*tw[k]+tb[k])
__device__ __forceinline__ v16h a_frag_time(float rt, const float* __restrict__ tw,
                                            const float* __restrict__ tb, int kloc, int half) {
  v16h a;
#pragma unroll
  for (int v = 0; v < 8; ++v) {
    int k = kloc + a_koff(v, half);
    a[2 * v]     = (_Float16)cosf(rt * tw[k]     + tb[k]);
    a[2 * v + 1] = (_Float16)cosf(rt * tw[k + 1] + tb[k + 1]);
  }
  return a;
}

// ---------- kernels ----------

__global__ void k_zero(int* counts, float* sums) {
  size_t i = (size_t)blockIdx.x * blockDim.x + threadIdx.x;
  if (i < NNODES) counts[i] = 0;
  if (i < (size_t)NNODES * MSG) sums[i] = 0.0f;
}

__global__ void k_count(const int* __restrict__ dst_ids, int* counts) {
  int i = blockIdx.x * blockDim.x + threadIdx.x;
  if (i < NE) atomicAdd(&counts[dst_ids[i]], 1);
}

// Single-block sorted-unique: presence scan over node ids (jnp.unique semantics)
__global__ void __launch_bounds__(1024) k_scan(const int* __restrict__ counts,
                                               int* id_of_rank, int* nuniq,
                                               float* out_unique) {
  __shared__ int part[1024];
  const int tid = threadIdx.x;
  const int CH = 20;             // 1024*20 >= 20000
  int base = tid * CH;
  int s = 0;
  for (int j = 0; j < CH; ++j) {
    int i = base + j;
    if (i < NNODES) s += (counts[i] > 0) ? 1 : 0;
  }
  part[tid] = s;
  __syncthreads();
  for (int off = 1; off < 1024; off <<= 1) {
    int v = part[tid];
    int o = (tid >= off) ? part[tid - off] : 0;
    __syncthreads();
    part[tid] = v + o;
    __syncthreads();
  }
  int r = (tid > 0) ? part[tid - 1] : 0;   // exclusive prefix
  if (tid == 1023) nuniq[0] = part[1023];
  for (int j = 0; j < CH; ++j) {
    int i = base + j;
    if (i < NNODES && counts[i] > 0) {
      id_of_rank[r] = i;
      out_unique[r] = (float)i;
      ++r;
    }
  }
}

__global__ void k_pad(const int* __restrict__ nuniq, int* id_of_rank, float* out_unique) {
  int r = blockIdx.x * blockDim.x + threadIdx.x;
  if (r < NNODES && r >= nuniq[0]) {
    id_of_rank[r] = -1;
    out_unique[r] = 0.0f;      // fill_value = 0
  }
}

// Pre-swizzle an f32 weight (K x Ncols, row-major) into per-lane contiguous
// B-matrix 32x16 f16 fragments: index ((kt*NT+nt)*32+lane)*16 + i,
// k = kt*32 + (i&7) + 8*(lane>>4) + 16*(i>>3), n = nt*16 + (lane&15)
__global__ void k_cvt_w(const float* __restrict__ W, _Float16* out,
                        int NT, int KT, int Ncols) {
  int idx = blockIdx.x * blockDim.x + threadIdx.x;
  int total = KT * NT * 512;
  if (idx >= total) return;
  int i = idx & 15;
  int lane = (idx >> 4) & 31;
  int t = idx >> 9;
  int nt = t % NT;
  int kt = t / NT;
  int half = lane >> 4, lm = lane & 15;
  int n = nt * 16 + lm;
  int k = kt * 32 + (i & 7) + 8 * half + 16 * (i >> 3);
  out[idx] = (_Float16)W[(size_t)k * Ncols + n];
}

// msg = relu(feats @ W_msg + b_msg); atomically accumulate per dst node id.
// 1 wave = 16 edges x 256 cols; 22 K-steps, 16 N-tiles resident (acc[16]).
__global__ void __launch_bounds__(256) k_msg(
    const float* __restrict__ rel, const float* __restrict__ srcm,
    const float* __restrict__ dstm, const float* __restrict__ ef,
    const int* __restrict__ dst_ids, const float* __restrict__ tw,
    const float* __restrict__ tb, const _Float16* __restrict__ wmsg_h,
    const float* __restrict__ bmsg, float* sums) {
  int lane = threadIdx.x & 31;
  int wave = threadIdx.x >> 5;
  int tile = blockIdx.x * 8 + wave;
  if (tile >= MTILES_E) return;
  int half = lane >> 4, lm = lane & 15;
  int e = tile * 16 + lm;                 // A row (lanes 0-15 & 16-31 same rows)
  float rt = rel[e];

  const v8f z8 = {0.f, 0.f, 0.f, 0.f, 0.f, 0.f, 0.f, 0.f};
  v8f acc[16];
#pragma unroll
  for (int i = 0; i < 16; ++i) acc[i] = z8;

#define MM_STEP(AF, KTG)                                                            \
  {                                                                                 \
    _Pragma("unroll") for (int nt = 0; nt < 16; ++nt) {                             \
      v16h b = *(const v16h*)(wmsg_h + ((size_t)((KTG) * 16 + nt) * 32 + lane) * 16); \
      acc[nt] = WMMA_F16(AF, b, acc[nt]);                                           \
    }                                                                               \
  }

  const float* srow = srcm + (size_t)e * MEM;
  const float* drow = dstm + (size_t)e * MEM;
  const float* erow = ef   + (size_t)e * EDGE;
  for (int kt = 0; kt < 8; ++kt) { v16h a = a_frag_f32(srow, kt * 32, half); MM_STEP(a, kt); }
  for (int kt = 0; kt < 8; ++kt) { v16h a = a_frag_f32(drow, kt * 32, half); MM_STEP(a, 8 + kt); }
  for (int kt = 0; kt < 4; ++kt) { v16h a = a_frag_f32(erow, kt * 32, half); MM_STEP(a, 16 + kt); }
  for (int kt = 0; kt < 2; ++kt) { v16h a = a_frag_time(rt, tw, tb, kt * 32, half); MM_STEP(a, 20 + kt); }
#undef MM_STEP

  // D layout: lane col = lm, row = r + 8*half (r = vgpr index)
  int did[8];
#pragma unroll
  for (int r = 0; r < 8; ++r) did[r] = dst_ids[tile * 16 + r + 8 * half];

#pragma unroll
  for (int nt = 0; nt < 16; ++nt) {
    int col = nt * 16 + lm;
    float bm = bmsg[col];
#pragma unroll
    for (int r = 0; r < 8; ++r) {
      float mv = acc[nt][r] + bm;
      if (mv > 0.0f) atomicAdd(&sums[(size_t)did[r] * MSG + col], mv);
    }
  }
}

// GRU: gx = agg@W_ih + b_ih, gh = prev@W_hh + b_hh, gate math, write memories.
// 1 wave = 16 rows; A fragments (agg + prev, 8 K-steps each) resident in VGPRs.
__global__ void __launch_bounds__(256) k_gru(
    const float* __restrict__ sums, const int* __restrict__ counts,
    const int* __restrict__ id_of_rank, const float* __restrict__ dstm,
    const _Float16* __restrict__ wih, const _Float16* __restrict__ whh,
    const float* __restrict__ bih, const float* __restrict__ bhh,
    float* __restrict__ outmem) {
  int lane = threadIdx.x & 31;
  int wave = threadIdx.x >> 5;
  int mt = blockIdx.x * 8 + wave;
  if (mt >= MTILES_N) return;
  int half = lane >> 4, lm = lane & 15;
  int m = mt * 16 + lm;                    // A row
  int id = id_of_rank[m];
  float invc = 0.0f;
  const float* srow = sums;
  if (id >= 0) {
    invc = 1.0f / fmaxf((float)counts[id], 1.0f);
    srow = sums + (size_t)id * MSG;
  }
  const float* prow = dstm + (size_t)m * MEM;

  v16h Ax[8], Ah[8];
#pragma unroll
  for (int kt = 0; kt < 8; ++kt) {
#pragma unroll
    for (int v = 0; v < 8; ++v) {
      int k = kt * 32 + a_koff(v, half);
      float a0 = (id >= 0) ? srow[k] * invc     : 0.0f;
      float a1 = (id >= 0) ? srow[k + 1] * invc : 0.0f;
      Ax[kt][2 * v]     = (_Float16)a0;
      Ax[kt][2 * v + 1] = (_Float16)a1;
      float2 p = *(const float2*)(prow + k);
      Ah[kt][2 * v]     = (_Float16)p.x;
      Ah[kt][2 * v + 1] = (_Float16)p.y;
    }
  }

  const v8f z8 = {0.f, 0.f, 0.f, 0.f, 0.f, 0.f, 0.f, 0.f};
  for (int nt = 0; nt < 16; ++nt) {
    v8f ax0 = z8, ax1 = z8, ax2 = z8, ah0 = z8, ah1 = z8, ah2 = z8;
#pragma unroll
    for (int kt = 0; kt < 8; ++kt) {
      v16h b;
      b = *(const v16h*)(wih + (((size_t)kt * 48 + nt     ) * 32 + lane) * 16); ax0 = WMMA_F16(Ax[kt], b, ax0);
      b = *(const v16h*)(wih + (((size_t)kt * 48 + nt + 16) * 32 + lane) * 16); ax1 = WMMA_F16(Ax[kt], b, ax1);
      b = *(const v16h*)(wih + (((size_t)kt * 48 + nt + 32) * 32 + lane) * 16); ax2 = WMMA_F16(Ax[kt], b, ax2);
      b = *(const v16h*)(whh + (((size_t)kt * 48 + nt     ) * 32 + lane) * 16); ah0 = WMMA_F16(Ah[kt], b, ah0);
      b = *(const v16h*)(whh + (((size_t)kt * 48 + nt + 16) * 32 + lane) * 16); ah1 = WMMA_F16(Ah[kt], b, ah1);
      b = *(const v16h*)(whh + (((size_t)kt * 48 + nt + 32) * 32 + lane) * 16); ah2 = WMMA_F16(Ah[kt], b, ah2);
    }
    int col = nt * 16 + lm;
    float bi0 = bih[col], bi1 = bih[col + 256], bi2 = bih[col + 512];
    float bh0 = bhh[col], bh1 = bhh[col + 256], bh2 = bhh[col + 512];
#pragma unroll
    for (int r = 0; r < 8; ++r) {
      int row = mt * 16 + r + 8 * half;
      float prev = dstm[(size_t)row * MEM + col];
      float xr = ax0[r] + bi0, xz = ax1[r] + bi1, xn = ax2[r] + bi2;
      float hr = ah0[r] + bh0, hz = ah1[r] + bh1, hn = ah2[r] + bh2;
      float rg = 1.0f / (1.0f + __expf(-(xr + hr)));
      float zg = 1.0f / (1.0f + __expf(-(xz + hz)));
      float ng = tanhf(xn + rg * hn);
      outmem[(size_t)row * MEM + col] = (1.0f - zg) * ng + zg * prev;
    }
  }
}

// ---------- launcher ----------

extern "C" void kernel_launch(void* const* d_in, const int* in_sizes, int n_in,
                              void* d_out, int out_size, void* d_ws, size_t ws_size,
                              hipStream_t stream) {
  const float* rel   = (const float*)d_in[0];
  const float* srcm  = (const float*)d_in[1];
  const float* dstm  = (const float*)d_in[2];
  const float* ef    = (const float*)d_in[3];
  const int*   dsti  = (const int*)  d_in[4];
  const float* tw    = (const float*)d_in[5];
  const float* tb    = (const float*)d_in[6];
  const float* Wmsg  = (const float*)d_in[7];
  const float* bmsg  = (const float*)d_in[8];
  const float* Wih   = (const float*)d_in[9];
  const float* Whh   = (const float*)d_in[10];
  const float* bih   = (const float*)d_in[11];
  const float* bhh   = (const float*)d_in[12];

  float* out        = (float*)d_out;
  float* out_unique = out;            // 20000 values (ids as float)
  float* out_mem    = out + NNODES;   // 20000 x 256 memories

  // workspace layout (all offsets 128B-aligned)
  char* W = (char*)d_ws;
  int*      counts     = (int*)(W);                    // 20000 ints
  int*      id_of_rank = (int*)(W + 81920);            // 20000 ints
  int*      nuniq      = (int*)(W + 163840);           // 1 int
  float*    sums       = (float*)(W + 164864);         // 20000*256 f32 = 20.48 MB
  _Float16* wmsg_h     = (_Float16*)(W + 20644864);    // 22*16*512 f16
  _Float16* wih_h      = (_Float16*)(W + 21005312);    // 8*48*512 f16
  _Float16* whh_h      = (_Float16*)(W + 21398528);    // 8*48*512 f16

  k_zero<<<(NNODES * MSG + 255) / 256, 256, 0, stream>>>(counts, sums);
  k_count<<<(NE + 255) / 256, 256, 0, stream>>>(dsti, counts);
  k_scan<<<1, 1024, 0, stream>>>(counts, id_of_rank, nuniq, out_unique);
  k_pad<<<(NNODES + 255) / 256, 256, 0, stream>>>(nuniq, id_of_rank, out_unique);

  k_cvt_w<<<(22 * 16 * 512 + 255) / 256, 256, 0, stream>>>(Wmsg, wmsg_h, 16, 22, MSG);
  k_cvt_w<<<(8 * 48 * 512 + 255) / 256, 256, 0, stream>>>(Wih, wih_h, 48, 8, G3);
  k_cvt_w<<<(8 * 48 * 512 + 255) / 256, 256, 0, stream>>>(Whh, whh_h, 48, 8, G3);

  k_msg<<<(MTILES_E + 7) / 8, 256, 0, stream>>>(rel, srcm, dstm, ef, dsti, tw, tb,
                                                wmsg_h, bmsg, sums);
  k_gru<<<(MTILES_N + 7) / 8, 256, 0, stream>>>(sums, counts, id_of_rank, dstm,
                                                wih_h, whh_h, bih, bhh, out_mem);
}
